// Attention_28845000360620
// MI455X (gfx1250) — matrix operand order repaired and checked by reading, not compiled, not run on hardware
//
#include <hip/hip_runtime.h>
#include <hip/hip_bf16.h>

// ---------------------------------------------------------------------------
// Multi-head attention forward for gfx1250 (MI455X).
// All matmuls via v_wmma_f32_16x16x32_f16 (fp16 operands, fp32 accum).
// Operand tiles staged in LDS with double-buffered GLOBAL_LOAD_ASYNC_TO_LDS
// (runtime pointer ping-pong so accumulators stay register-resident),
// drained with s_wait_asynccnt.
// Shapes: BATCH=2, SEQ=2048, DIM=1024, N_HEADS=16, HEAD_DIM=64, hd=1024.
// ---------------------------------------------------------------------------

typedef _Float16 v16h __attribute__((ext_vector_type(16)));
typedef _Float16 v8h  __attribute__((ext_vector_type(8)));
typedef float    v8f  __attribute__((ext_vector_type(8)));

#define BATCH    2
#define SEQ      2048
#define DIM      1024
#define NHEADS   16
#define HDIM     64
#define HD       1024            // NHEADS*HDIM
#define TOKENS   (BATCH*SEQ)     // 4096

__device__ __forceinline__ v8f wmma_f16(v16h a, v16h b, v8f c) {
  return __builtin_amdgcn_wmma_f32_16x16x32_f16(
      /*neg_a=*/false, a, /*neg_b=*/false, b,
      /*c_mod=*/(short)0, c, /*reuse_a=*/false, /*reuse_b=*/false);
}

// A-fragment (16x32 f16): per lane, 16 contiguous halves at p (p and p+8).
__device__ __forceinline__ v16h load_a16(const _Float16* p) {
  v8h lo = *(const v8h*)p;
  v8h hi = *(const v8h*)(p + 8);
  return __builtin_shufflevector(lo, hi, 0,1,2,3,4,5,6,7,8,9,10,11,12,13,14,15);
}

// Async copy 16 bytes global -> LDS (per lane). Tracked by ASYNCcnt.
// Low 32 bits of a generic shared pointer are the LDS byte offset.
__device__ __forceinline__ void async_b128(const _Float16* gsrc,
                                           const _Float16* ldst) {
  unsigned l = (unsigned)(uintptr_t)ldst;
  asm volatile("global_load_async_to_lds_b128 %0, %1, off"
               :: "v"(l), "v"(gsrc) : "memory");
}
#define WAIT_ASYNC(n) asm volatile("s_wait_asynccnt " #n ::: "memory")

// ---------------------------------------------------------------------------
// Prep: f32 -> f16 (vectorized x4)
// ---------------------------------------------------------------------------
__global__ void cvt_f32_to_f16(const float* __restrict__ src,
                               _Float16* __restrict__ dst, int n) {
  int i = (blockIdx.x * blockDim.x + threadIdx.x) * 4;
  if (i + 3 < n) {
    float4 v = *(const float4*)(src + i);
    dst[i + 0] = (_Float16)v.x;
    dst[i + 1] = (_Float16)v.y;
    dst[i + 2] = (_Float16)v.z;
    dst[i + 3] = (_Float16)v.w;
  }
}

// Prep: transpose-convert weight (rows x cols f32) -> dst[n*rows+k]
__global__ void cvt_transpose_f16(const float* __restrict__ src,
                                  _Float16* __restrict__ dst,
                                  int rows, int cols) {
  int idx = blockIdx.x * blockDim.x + threadIdx.x;
  if (idx >= rows * cols) return;
  int n = idx / rows;
  int k = idx - n * rows;
  dst[idx] = (_Float16)src[k * cols + n];
}

// ---------------------------------------------------------------------------
// Fused QKV projection: C = xh(4096x1024) @ [Wq|Wk|Wv] (N_total=3072).
// Block tile: 256(M) x 64(N); 8 waves each own 32(M)x64(N) (2 A-frags,
// 8 WMMA per k-step) and share one async-staged LDS B tile (64x32 halves,
// double buffered via pointer ping-pong).
// Q,K stored (b,h,s,d); V stored transposed (b,h,d,s).
// ---------------------------------------------------------------------------
__global__ __launch_bounds__(256) void qkv_gemm(
    const _Float16* __restrict__ xh, const _Float16* __restrict__ wT,
    _Float16* __restrict__ q, _Float16* __restrict__ k,
    _Float16* __restrict__ vt) {
  __shared__ __align__(32) _Float16 Bb[2][64][32];

  const int tid  = threadIdx.x;
  const int wib  = tid >> 5;
  const int lane = tid & 31;
  const int hi   = lane >> 4;
  const int ln   = lane & 15;
  const int blk_m    = blockIdx.x / 48;           // 0..15  (4096/256)
  const int col_grp  = blockIdx.x - blk_m * 48;   // 0..47  (3072/64)
  const int m0 = blk_m * 256 + wib * 32;
  const int n0 = col_grp * 64;

  // async-stage slice for this thread: one b128 per fill
  const int fill_off = (tid >> 2) * 32 + (tid & 3) * 8;
  const _Float16* bsrc = wT + (size_t)(n0 + (tid >> 2)) * DIM + (tid & 3) * 8;

  const _Float16* bufc = &Bb[0][0][0];
  const _Float16* bufn = &Bb[1][0][0];

  v8f accA[4] = {}, accB[4] = {};
  const _Float16* arowA = xh + (size_t)(m0 + ln) * DIM + hi * 16;
  const _Float16* arowB = arowA + (size_t)16 * DIM;

  async_b128(bsrc, bufn - 2048 + fill_off);     // fill Bb[0] (bufc)
#pragma unroll 1
  for (int i = 0; i < DIM / 32; ++i) {
    const int k0 = i * 32;
    if (i < DIM / 32 - 1) {
      async_b128(bsrc + k0 + 32, bufn + fill_off);
      WAIT_ASYNC(1);                  // in-order: current tile complete
    } else {
      WAIT_ASYNC(0);
    }
    __syncthreads();                  // all waves' fills visible

    v16h aA = load_a16(arowA + k0);
    v16h aB = load_a16(arowB + k0);
#pragma unroll
    for (int t = 0; t < 4; ++t) {
      v16h b = *(const v16h*)(bufc + (t * 16 + ln) * 32 + hi * 16);
      accA[t] = wmma_f16(aA, b, accA[t]);
      accB[t] = wmma_f16(aB, b, accB[t]);
    }
    __syncthreads();                  // protect buffer overwrite
    const _Float16* tsw = bufc; bufc = bufn; bufn = tsw;
  }

  // Route results: section 0 -> Q, 1 -> K, 2 -> V^T
  const int sec = n0 >> 10;
  const int nn0 = n0 & 1023;
  const int h   = nn0 >> 6;
  const int b   = m0 >> 11;
  const int s0  = m0 & 2047;
  const int bh  = b * NHEADS + h;
#pragma unroll
  for (int t = 0; t < 4; ++t) {
    const int d = t * 16 + ln;
#pragma unroll
    for (int v = 0; v < 8; ++v) {
#pragma unroll
      for (int g = 0; g < 2; ++g) {
        const int r = g * 16 + v + hi * 8;
        _Float16 val = (_Float16)(g == 0 ? accA[t][v] : accB[t][v]);
        if (sec == 2) {
          vt[((size_t)bh * HDIM + d) * SEQ + (s0 + r)] = val;
        } else {
          _Float16* dst = (sec == 0) ? q : k;
          dst[((size_t)bh * SEQ + (s0 + r)) * HDIM + d] = val;
        }
      }
    }
  }
}

// ---------------------------------------------------------------------------
// Flash attention: 8 waves per block share one (b,h); each wave owns a
// 16-query block. K (32x64) and V^T (64x32) tiles double-buffered in LDS
// via async loads (pointer ping-pong); online softmax per wave through LDS.
// ---------------------------------------------------------------------------
__global__ __launch_bounds__(256) void attn_kernel(
    const _Float16* __restrict__ q, const _Float16* __restrict__ kmat,
    const _Float16* __restrict__ vt, _Float16* __restrict__ a16) {
  __shared__ __align__(32) _Float16 Kb[2][32][64];     // 8 KB
  __shared__ __align__(32) _Float16 Vb[2][64][32];     // 8 KB
  __shared__ __align__(32) float    sc[8][16 * 32];    // 16 KB
  __shared__ __align__(32) _Float16 pt[8][16 * 32];    // 8 KB
  __shared__ float mrow[8][16], lrow[8][16], frow[8][16];

  const int tid  = threadIdx.x;
  const int wib  = tid >> 5;
  const int lane = tid & 31;
  const int hi   = lane >> 4;
  const int ln   = lane & 15;
  const int bh   = blockIdx.x >> 4;                 // 0..31, shared by block
  const int q0   = ((blockIdx.x & 15) * 8 + wib) * 16;

  const _Float16* qp = q    + (size_t)bh * SEQ * HDIM;
  const _Float16* kp = kmat + (size_t)bh * SEQ * HDIM;
  const _Float16* vp = vt   + (size_t)bh * HDIM * SEQ;

  // async-stage slices for this thread (one b128 each for K and V tiles)
  const int kfill = (tid >> 3) * 64 + (tid & 7) * 8;   // into Kb[buf]
  const int vfill = (tid >> 2) * 32 + (tid & 3) * 8;   // into Vb[buf]
  const _Float16* ksrc = kp + (size_t)(tid >> 3) * HDIM + (tid & 7) * 8;
  const _Float16* vsrc = vp + (size_t)(tid >> 2) * SEQ + (tid & 3) * 8;

  const _Float16* kbc = &Kb[0][0][0];
  const _Float16* kbn = &Kb[1][0][0];
  const _Float16* vbc = &Vb[0][0][0];
  const _Float16* vbn = &Vb[1][0][0];

  // Q fragments for d=0..31 and d=32..63
  const _Float16* qrow = qp + (size_t)(q0 + ln) * HDIM + hi * 16;
  v16h a0 = load_a16(qrow);
  v16h a1 = load_a16(qrow + 32);

  if (lane < 16) {
    mrow[wib][lane] = -3.0e38f;
    lrow[wib][lane] = 0.0f;
  }

  v8f acc[4] = {};
  const float scale = 0.125f;  // 1/sqrt(64)

  async_b128(ksrc, kbc + kfill);               // prologue fills into buf 0
  async_b128(vsrc, vbc + vfill);
#pragma unroll 1
  for (int i = 0; i < SEQ / 32; ++i) {
    if (i < SEQ / 32 - 1) {
      async_b128(ksrc + (size_t)32 * HDIM, kbn + kfill);
      async_b128(vsrc + 32,                vbn + vfill);
      ksrc += (size_t)32 * HDIM;
      vsrc += 32;
      WAIT_ASYNC(2);                 // in-order: current K+V tiles complete
    } else {
      WAIT_ASYNC(0);
    }
    __syncthreads();

    // ---- scores: 16 queries x 32 keys, contraction over d=64 ----
    v8f s0 = {}, s1 = {};
    {
      v16h b00 = *(const v16h*)(kbc + ln * 64 + hi * 16);
      v16h b01 = *(const v16h*)(kbc + ln * 64 + 32 + hi * 16);
      v16h b10 = *(const v16h*)(kbc + (16 + ln) * 64 + hi * 16);
      v16h b11 = *(const v16h*)(kbc + (16 + ln) * 64 + 32 + hi * 16);
      s0 = wmma_f16(a0, b00, s0);
      s0 = wmma_f16(a1, b01, s0);
      s1 = wmma_f16(a0, b10, s1);
      s1 = wmma_f16(a1, b11, s1);
    }
    // stage scores to LDS (C layout: row = v + hi*8, col = ln)
#pragma unroll
    for (int v = 0; v < 8; ++v) {
      const int r = v + hi * 8;
      sc[wib][r * 32 + ln]      = s0[v];
      sc[wib][r * 32 + 16 + ln] = s1[v];
    }
    asm volatile("s_wait_dscnt 0" ::: "memory");

    // ---- online softmax: lanes 0..15 each own one query row ----
    if (lane < 16) {
      const int r = lane;
      float m_old = mrow[wib][r];
      float srow[32];
      float tmax = m_old;
#pragma unroll
      for (int j = 0; j < 32; ++j) {
        srow[j] = sc[wib][r * 32 + j] * scale;
        tmax = fmaxf(tmax, srow[j]);
      }
      float fac = __expf(m_old - tmax);
      float tsum = 0.0f;
#pragma unroll
      for (int j = 0; j < 32; ++j) {
        float p = __expf(srow[j] - tmax);
        tsum += p;
        pt[wib][r * 32 + j] = (_Float16)p;
      }
      mrow[wib][r] = tmax;
      lrow[wib][r] = lrow[wib][r] * fac + tsum;
      frow[wib][r] = fac;
    }
    asm volatile("s_wait_dscnt 0" ::: "memory");

    // ---- rescale accumulators by per-row factor ----
#pragma unroll
    for (int v = 0; v < 8; ++v) {
      float f = frow[wib][v + hi * 8];
      acc[0][v] *= f; acc[1][v] *= f; acc[2][v] *= f; acc[3][v] *= f;
    }

    // ---- P as A-fragment from LDS ----
    const _Float16* pp = &pt[wib][ln * 32 + hi * 16];
    v8h plo = *(const v8h*)pp;
    v8h phi = *(const v8h*)(pp + 8);
    v16h ap = __builtin_shufflevector(plo, phi,
                                      0,1,2,3,4,5,6,7,8,9,10,11,12,13,14,15);
    // ---- PV: accumulate 16x64 ----
#pragma unroll
    for (int t = 0; t < 4; ++t) {
      v16h bv = *(const v16h*)(vbc + (t * 16 + ln) * 32 + hi * 16);
      acc[t] = wmma_f16(ap, bv, acc[t]);
    }
    __syncthreads();                  // protect K/V buffer overwrite
    const _Float16* t0 = kbc; kbc = kbn; kbn = t0;
    const _Float16* t1 = vbc; vbc = vbn; vbn = t1;
  }

  // ---- epilogue: divide by row sums, store (token, h*64+d) f16 ----
  const int b = bh >> 4, h = bh & 15;
#pragma unroll
  for (int v = 0; v < 8; ++v) {
    const int r = v + hi * 8;
    const float invl = 1.0f / lrow[wib][r];
    const size_t token = (size_t)b * SEQ + q0 + r;
#pragma unroll
    for (int t = 0; t < 4; ++t) {
      a16[token * HD + h * HDIM + t * 16 + ln] = (_Float16)(acc[t][v] * invl);
    }
  }
}

// ---------------------------------------------------------------------------
// Output projection: out(f32) = a16(4096x1024) @ Wo + bo, via WoT (N,K).
// Same structure as qkv_gemm; block tile 256x64.
// ---------------------------------------------------------------------------
__global__ __launch_bounds__(256) void out_gemm(
    const _Float16* __restrict__ a16, const _Float16* __restrict__ woT,
    const float* __restrict__ bo, float* __restrict__ out) {
  __shared__ __align__(32) _Float16 Bb[2][64][32];

  const int tid  = threadIdx.x;
  const int wib  = tid >> 5;
  const int lane = tid & 31;
  const int hi   = lane >> 4;
  const int ln   = lane & 15;
  const int m0 = (blockIdx.x >> 4) * 256 + wib * 32;  // 16 M-blocks
  const int n0 = (blockIdx.x & 15) * 64;              // 16 N-groups

  const int fill_off = (tid >> 2) * 32 + (tid & 3) * 8;
  const _Float16* bsrc = woT + (size_t)(n0 + (tid >> 2)) * HD + (tid & 3) * 8;

  const _Float16* bufc = &Bb[0][0][0];
  const _Float16* bufn = &Bb[1][0][0];

  v8f accA[4] = {}, accB[4] = {};
  const _Float16* arowA = a16 + (size_t)(m0 + ln) * HD + hi * 16;
  const _Float16* arowB = arowA + (size_t)16 * HD;

  async_b128(bsrc, bufc + fill_off);
#pragma unroll 1
  for (int i = 0; i < HD / 32; ++i) {
    const int k0 = i * 32;
    if (i < HD / 32 - 1) {
      async_b128(bsrc + k0 + 32, bufn + fill_off);
      WAIT_ASYNC(1);
    } else {
      WAIT_ASYNC(0);
    }
    __syncthreads();

    v16h aA = load_a16(arowA + k0);
    v16h aB = load_a16(arowB + k0);
#pragma unroll
    for (int t = 0; t < 4; ++t) {
      v16h b = *(const v16h*)(bufc + (t * 16 + ln) * 32 + hi * 16);
      accA[t] = wmma_f16(aA, b, accA[t]);
      accB[t] = wmma_f16(aB, b, accB[t]);
    }
    __syncthreads();
    const _Float16* tsw = bufc; bufc = bufn; bufn = tsw;
  }

#pragma unroll
  for (int t = 0; t < 4; ++t) {
    const int n = n0 + t * 16 + ln;
    const float bias = bo[n];
#pragma unroll
    for (int v = 0; v < 8; ++v) {
      out[(size_t)(m0 + v + hi * 8) * DIM + n]      = accA[t][v] + bias;
      out[(size_t)(m0 + 16 + v + hi * 8) * DIM + n] = accB[t][v] + bias;
    }
  }
}

// ---------------------------------------------------------------------------
// Launch
// ---------------------------------------------------------------------------
extern "C" void kernel_launch(void* const* d_in, const int* in_sizes, int n_in,
                              void* d_out, int out_size, void* d_ws,
                              size_t ws_size, hipStream_t stream) {
  const float* x  = (const float*)d_in[0];
  const float* Wq = (const float*)d_in[1];
  const float* Wk = (const float*)d_in[2];
  const float* Wv = (const float*)d_in[3];
  const float* Wo = (const float*)d_in[4];
  const float* bo = (const float*)d_in[5];
  float* out = (float*)d_out;

  // Workspace carve-up (all f16): 24M halves = 48 MB
  _Float16* xh    = (_Float16*)d_ws;                  // 4096 x 1024
  _Float16* wqkvT = xh    + (size_t)TOKENS * DIM;     // 3072 x 1024
  _Float16* woT   = wqkvT + (size_t)3 * HD * DIM;     // 1024 x 1024
  _Float16* Q     = woT   + (size_t)DIM * HD;         // (b,h,s,d)
  _Float16* K     = Q     + (size_t)BATCH * NHEADS * SEQ * HDIM;
  _Float16* VT    = K     + (size_t)BATCH * NHEADS * SEQ * HDIM; // (b,h,d,s)
  _Float16* A16   = VT    + (size_t)BATCH * NHEADS * HDIM * SEQ; // 4096 x 1024

  // Prep
  cvt_f32_to_f16<<<(TOKENS * DIM / 4 + 255) / 256, 256, 0, stream>>>(
      x, xh, TOKENS * DIM);
  cvt_transpose_f16<<<(DIM * HD + 255) / 256, 256, 0, stream>>>(
      Wq, wqkvT, DIM, HD);
  cvt_transpose_f16<<<(DIM * HD + 255) / 256, 256, 0, stream>>>(
      Wk, wqkvT + (size_t)HD * DIM, DIM, HD);
  cvt_transpose_f16<<<(DIM * HD + 255) / 256, 256, 0, stream>>>(
      Wv, wqkvT + (size_t)2 * HD * DIM, DIM, HD);
  cvt_transpose_f16<<<(HD * DIM + 255) / 256, 256, 0, stream>>>(
      Wo, woT, HD, DIM);

  // QKV projection: 16 M-blocks x 48 N-groups, 256x64 per block
  qkv_gemm<<<768, 256, 0, stream>>>(xh, wqkvT, Q, K, VT);

  // Attention: 32 (b,h) x 16 q-groups, 8 waves/block sharing K/V tiles
  attn_kernel<<<512, 256, 0, stream>>>(Q, K, VT, A16);

  // Output projection: 16 M-blocks x 16 N-groups
  out_gemm<<<256, 256, 0, stream>>>(A16, woT, bo, out);
}
